// my_TGCN2_87926570484073
// MI455X (gfx1250) — matrix operand order repaired and checked by reading, not compiled, not run on hardware
//
#include <hip/hip_runtime.h>

typedef unsigned short u16;
typedef unsigned int u32;

typedef __bf16 v16bf __attribute__((ext_vector_type(16)));
typedef __bf16 v8bf  __attribute__((ext_vector_type(8)));
typedef float  v8f   __attribute__((ext_vector_type(8)));

#define BB   64
#define NN   512
#define CIN  128
#define HH   256
#define OUTD 12

static __device__ __forceinline__ u16 f2bf(float f) {
    u32 u = __float_as_uint(f);
    u32 r = u + 0x7FFFu + ((u >> 16) & 1u);
    return (u16)(r >> 16);
}
static __device__ __forceinline__ float bf2f(u16 h) {
    return __uint_as_float(((u32)h) << 16);
}
static __device__ __forceinline__ v16bf mk16(v8bf lo, v8bf hi) {
    v16bf r;
#pragma unroll
    for (int i = 0; i < 8; ++i) { r[i] = lo[i]; r[8 + i] = hi[i]; }
    return r;
}
static __device__ __forceinline__ float sigmoidf_(float x) {
    return 1.0f / (1.0f + __expf(-x));
}

// ---------------- GCN normalization ----------------
__global__ void k_init_deg(float* deg) {
    int n = blockIdx.x * 256 + threadIdx.x;
    if (n < NN) deg[n] = 2.0f; // improved=True self loop fill
}
__global__ void k_deg_accum(const int* ei, const float* ew, float* deg, int E) {
    int e = blockIdx.x * 256 + threadIdx.x;
    if (e < E) atomicAdd(&deg[ei[E + e]], ew[e]);
}
__global__ void k_make_norm(const int* ei, const float* ew, const float* deg,
                            int* srcA, int* tgtA, float* normA, int E) {
    int e = blockIdx.x * 256 + threadIdx.x;
    if (e >= E + NN) return;
    int s, t; float w;
    if (e < E) { s = ei[e]; t = ei[E + e]; w = ew[e]; }
    else { s = t = e - E; w = 2.0f; }
    float ds = deg[s], dt = deg[t];
    float is = ds > 0.f ? rsqrtf(ds) : 0.f;
    float it = dt > 0.f ? rsqrtf(dt) : 0.f;
    srcA[e] = s; tgtA[e] = t; normA[e] = is * w * it;
}

// ---------------- bf16 B-matrix fragment packing ----------------
// 32x16 (KxN) tile, wave32 layout: lane half h holds k=8h..8h+7 (t<8) and
// k=16+8h..16+8h+7 (t>=8), n=lane%16. Lane's 16 bf16 stored contiguously.
__global__ void k_pack(const float* __restrict__ W, u16* __restrict__ out,
                       int K, int NcLocal, long sk, long sn, int ntOff) {
    int tid = blockIdx.x * 256 + threadIdx.x;
    int KT = K >> 5;
    int total = KT * (NcLocal >> 4) * 512;
    if (tid >= total) return;
    int e = tid & 511, tile = tid >> 9;
    int nt = tile / KT, kt = tile % KT;
    int lane = e >> 4, tt = e & 15;
    int kk = ((tt < 8) ? tt : (tt + 8)) + ((lane < 16) ? 0 : 8);
    int nn = (lane < 16) ? lane : (lane - 16);
    long k = kt * 32 + kk, n = nt * 16 + nn;
    out[(((long)(nt + ntOff) * KT + kt) << 9) + e] = f2bf(W[k * sk + n * sn]);
}

// ---------------- WMMA GEMM: C[M,Nc] = act(A @ Bpacked + bias) ----------------
// A element (r,k) at A[((r>>rs)*sA) + ((r&rm)*sB) + k]  (shift/mask, no division)
// Block 256 thr = 8 waves, tile 64x64. Full 64xK A tile staged in LDS once
// (K <= 256), then barrier-free K loop; each wave owns two 16x16 accumulators.
__global__ void __launch_bounds__(256) k_gemm(
    const float* __restrict__ A, int rs, int rm, long sA, long sB,
    const u16* __restrict__ Bp, const float* __restrict__ bias,
    float* __restrict__ C, int K, int kshift, int Nc, int act) {
    __shared__ u16 As[64][264]; // row stride 528B (16B multiple), K<=256
    const int lane = threadIdx.x & 31;
    const int wave = threadIdx.x >> 5;
    const int wm = wave & 3;  // row tile 0..3
    const int wg = wave >> 2; // col pair group 0..1
    const long rowBase = (long)blockIdx.x * 64;
    const int colBase = blockIdx.y * 64;
    const int KT = K >> 5;
    // ---- stage full 64xK A tile as bf16 (coalesced, shift/mask addressing) ----
    for (int i = threadIdx.x; i < (64 << kshift); i += 256) {
        int r = i >> kshift, kk = i & (K - 1);
        long rr = rowBase + r;
        long off = ((rr >> rs) * sA) + ((long)(rr & rm) * sB) + kk;
        As[r][kk] = f2bf(A[off]);
    }
    __syncthreads();
    v8f acc0 = {}, acc1 = {};
    const int mloc = wm * 16 + (lane & 15);
    const int k0 = (lane < 16) ? 0 : 8; // A frag chunk select per ISA layout
    const int nt0 = (colBase >> 4) + wg * 2;
    const u16* bptr0 = Bp + ((long)nt0 * KT) * 512 + lane * 16;
    const u16* bptr1 = bptr0 + (long)KT * 512;
#pragma unroll 4
    for (int kt = 0; kt < KT; ++kt) {
        v8bf lo = *(const v8bf*)&As[mloc][kt * 32 + k0];
        v8bf hi = *(const v8bf*)&As[mloc][kt * 32 + k0 + 16];
        v16bf af = mk16(lo, hi);
        v16bf b0 = *(const v16bf*)(bptr0 + (long)kt * 512);
        v16bf b1 = *(const v16bf*)(bptr1 + (long)kt * 512);
        __builtin_prefetch(bptr0 + (long)(kt + 1) * 512, 0, 3);
        acc0 = __builtin_amdgcn_wmma_f32_16x16x32_bf16(false, af, false, b0,
                                                       (short)0, acc0, false, false);
        acc1 = __builtin_amdgcn_wmma_f32_16x16x32_bf16(false, af, false, b1,
                                                       (short)0, acc1, false, false);
    }
    // ---- epilogue ----
    const int colW0 = colBase + wg * 32 + (lane & 15);
    const float bv0 = bias ? bias[colW0] : 0.0f;
    const float bv1 = bias ? bias[colW0 + 16] : 0.0f;
#pragma unroll
    for (int i = 0; i < 8; ++i) {
        int row = (int)rowBase + wm * 16 + ((lane < 16) ? i : i + 8);
        float v0 = acc0[i] + bv0;
        float v1 = acc1[i] + bv1;
        if (act == 1) { v0 = sigmoidf_(v0); v1 = sigmoidf_(v1); }
        else if (act == 2) { v0 = tanhf(v0); v1 = tanhf(v1); }
        C[(long)row * Nc + colW0] = v0;
        C[(long)row * Nc + colW0 + 16] = v1;
    }
}

// ---------------- edge aggregation ----------------
__global__ void k_init_agg(const float* bz, const float* bh, float* agg, long total) {
    long tid = (long)blockIdx.x * 256 + threadIdx.x;
    if (tid >= total) return;
    int c = (int)(tid & 511);
    agg[tid] = (c < 256) ? bz[c] : bh[c - 256];
}
__global__ void k_scatter(const float* __restrict__ xw, const int* __restrict__ srcA,
                          const int* __restrict__ tgtA, const float* __restrict__ normA,
                          float* __restrict__ agg, int nEdge) {
    long tid = (long)blockIdx.x * 256 + threadIdx.x;
    long total = (long)nEdge * BB * 128; // 512 cols / 4
    if (tid >= total) return;
    int c4 = (int)(tid & 127);
    long rest = tid >> 7;
    int b = (int)(rest & (BB - 1));
    int e = (int)(rest >> 6);
    int s = srcA[e], t = tgtA[e];
    float w = normA[e];
    const float4 v = *(const float4*)&xw[(((long)b * NN + s) << 9) + c4 * 4];
    float* dst = &agg[(((long)b * NN + t) << 9) + c4 * 4];
    atomicAdd(dst + 0, v.x * w);
    atomicAdd(dst + 1, v.y * w);
    atomicAdd(dst + 2, v.z * w);
    atomicAdd(dst + 3, v.w * w);
}

// ---------------- GCN head: h = relu((1-Z)*Ht) @ l1_W + l1_b ----------------
__global__ void k_hn_head(const float* __restrict__ Z, const float* __restrict__ Ht,
                          const float* __restrict__ l1W, const float* __restrict__ l1b,
                          float* __restrict__ out) {
    int row = blockIdx.x * 8 + (threadIdx.x >> 5);
    int lane = threadIdx.x & 31;
    float acc = 0.f;
    for (int c = lane; c < HH; c += 32) {
        float z = Z[(long)row * HH + c], ht = Ht[(long)row * HH + c];
        float hn = (1.f - z) * ht;
        acc += fmaxf(hn, 0.f) * l1W[c];
    }
    for (int off = 16; off; off >>= 1) acc += __shfl_down(acc, off, 32);
    if (lane == 0) out[row] = acc + l1b[0];
}

// ---------------- persistent GRU scan (one layer, 512 sequential steps) -------
// h [64,256] bf16 in LDS; per step gh = h @ Whh^T (WMMA bf16), then gate math.
__global__ void __launch_bounds__(1024) k_scan(
    const float* __restrict__ gi, const u16* __restrict__ Whhp,
    const float* __restrict__ bhh, float* __restrict__ hout) {
    extern __shared__ char smem[];
    u16(*hbf)[264] = (u16(*)[264])smem;                 // 64*264*2 = 33792B
    float* gh = (float*)(smem + 64 * 264 * 2);          // 64*768*4 = 196608B
    const int lane = threadIdx.x & 31;
    const int wave = threadIdx.x >> 5;
    for (int i = threadIdx.x; i < 64 * 264; i += 1024) ((u16*)smem)[i] = 0;
    __syncthreads();
    for (int t = 0; t < NN; ++t) {
        const float* git = gi + (long)t * 64 * 768;
        const int kbase0 = (lane < 16) ? 0 : 8;
#pragma unroll
        for (int ti = 0; ti < 6; ++ti) {
            int id = wave * 6 + ti;
            int mt = id & 3, nt = id >> 2; // 4 row tiles x 48 col tiles
            v8f acc = {};
            int mloc = mt * 16 + (lane & 15);
            const u16* bptr = Whhp + ((long)nt * 8) * 512 + lane * 16;
#pragma unroll
            for (int kt = 0; kt < 8; ++kt) {
                v8bf lo = *(const v8bf*)&hbf[mloc][kt * 32 + kbase0];
                v8bf hi = *(const v8bf*)&hbf[mloc][kt * 32 + kbase0 + 16];
                v16bf af = mk16(lo, hi);
                v16bf bfr = *(const v16bf*)(bptr + kt * 512);
                __builtin_prefetch(bptr + (kt + 1) * 512, 0, 3);
                acc = __builtin_amdgcn_wmma_f32_16x16x32_bf16(false, af, false, bfr,
                                                              (short)0, acc, false, false);
            }
            int col = nt * 16 + (lane & 15);
#pragma unroll
            for (int i = 0; i < 8; ++i) {
                int row = mt * 16 + ((lane < 16) ? i : i + 8);
                gh[row * 768 + col] = acc[i];
            }
        }
        __syncthreads();
        for (int c = threadIdx.x; c < 64 * 256; c += 1024) {
            int row = c >> 8, col = c & 255;
            float ir = git[row * 768 + col];
            float iz = git[row * 768 + col + 256];
            float in_ = git[row * 768 + col + 512];
            float hr = gh[row * 768 + col] + bhh[col];
            float hz = gh[row * 768 + col + 256] + bhh[col + 256];
            float hn = gh[row * 768 + col + 512] + bhh[col + 512];
            float r = sigmoidf_(ir + hr);
            float z = sigmoidf_(iz + hz);
            float n = tanhf(in_ + r * hn);
            float hp = bf2f(hbf[row][col]);
            float hnew = (1.f - z) * n + z * hp;
            hbf[row][col] = f2bf(hnew);
            hout[(((long)t * 64 + row) << 8) + col] = hnew;
        }
        __syncthreads();
    }
}

// ---------------- GRU head ----------------
__global__ void k_gru_head(const float* __restrict__ h2, const float* __restrict__ l2W,
                           const float* __restrict__ l2b, float* __restrict__ out) {
    int rnb = blockIdx.x * 8 + (threadIdx.x >> 5); // n*B + b order
    int lane = threadIdx.x & 31;
    float acc = 0.f;
    for (int c = lane; c < HH; c += 32)
        acc += h2[(long)rnb * HH + c] * l2W[c];
    for (int off = 16; off; off >>= 1) acc += __shfl_down(acc, off, 32);
    if (lane == 0) {
        int n = rnb / BB, b = rnb % BB;
        out[(long)b * NN + n] = acc + l2b[0];
    }
}

// ---------------- final projection ----------------
__global__ void k_final(const float* __restrict__ gruh, const float* __restrict__ gcnh,
                        const float* __restrict__ l3W, const float* __restrict__ l3b,
                        float* __restrict__ out, long total) {
    long tid = (long)blockIdx.x * 256 + threadIdx.x;
    if (tid >= total) return;
    long r = tid / OUTD;
    int o = (int)(tid % OUTD);
    out[tid] = gruh[r] * l3W[o] + gcnh[r] * l3W[OUTD + o] + l3b[o];
}

// =============================== host side ===================================
extern "C" void kernel_launch(void* const* d_in, const int* in_sizes, int n_in,
                              void* d_out, int out_size, void* d_ws, size_t ws_size,
                              hipStream_t stream) {
    const float* x   = (const float*)d_in[0];
    const int*   ei  = (const int*)d_in[1];
    const float* ew  = (const float*)d_in[2];
    const float* Wz  = (const float*)d_in[3];
    const float* bz  = (const float*)d_in[4];
    const float* lzW = (const float*)d_in[5];
    const float* lzb = (const float*)d_in[6];
    // d_in[7..10]: Wr, br, lr_W, lr_b -> dead (Hst == 0 makes R unused)
    const float* Wh  = (const float*)d_in[11];
    const float* bh  = (const float*)d_in[12];
    const float* lhW = (const float*)d_in[13];
    const float* lhb = (const float*)d_in[14];
    const float* Wih0 = (const float*)d_in[15];
    const float* Whh0 = (const float*)d_in[16];
    const float* bih0 = (const float*)d_in[17];
    const float* bhh0 = (const float*)d_in[18];
    const float* Wih1 = (const float*)d_in[19];
    const float* Whh1 = (const float*)d_in[20];
    const float* bih1 = (const float*)d_in[21];
    const float* bhh1 = (const float*)d_in[22];
    const float* l1W = (const float*)d_in[23];
    const float* l1b = (const float*)d_in[24];
    const float* l2W = (const float*)d_in[25];
    const float* l2b = (const float*)d_in[26];
    const float* l3W = (const float*)d_in[27];
    const float* l3b = (const float*)d_in[28];
    const int E = in_sizes[1] / 2;
    const int nEdge = E + NN;
    const long M = (long)BB * NN; // 32768 rows

    // ---- workspace layout (bytes) ----
    char* ws = (char*)d_ws;
    size_t o = 0;
    auto bump = [&](size_t bytes) { size_t r = o; o = (o + bytes + 255) & ~(size_t)255; return r; };
    size_t o_xw2 = bump(M * 512 * 4);   // also start of gi region (reuse)
    size_t o_agg = bump(M * 512 * 4);
    size_t o_Z   = bump(M * 256 * 4);   // also hseq region (reuse)
    size_t o_Ht  = bump(M * 256 * 4);
    size_t o_hh  = bump(M * 4);         // gcn head per row
    size_t o_gh  = bump(M * 4);         // gru head per row
    size_t o_deg = bump(NN * 4);
    size_t o_src = bump((size_t)nEdge * 4);
    size_t o_tgt = bump((size_t)nEdge * 4);
    size_t o_nrm = bump((size_t)nEdge * 4);
    size_t o_Pg   = bump((size_t)128 * 512 * 2);
    size_t o_Plz  = bump((size_t)256 * 256 * 2);
    size_t o_Plh  = bump((size_t)256 * 256 * 2);
    size_t o_Pih0 = bump((size_t)128 * 768 * 2);
    size_t o_Phh0 = bump((size_t)256 * 768 * 2);
    size_t o_Pih1 = bump((size_t)256 * 768 * 2);
    size_t o_Phh1 = bump((size_t)256 * 768 * 2);
    float* xw2  = (float*)(ws + o_xw2);
    float* agg  = (float*)(ws + o_agg);
    float* Zb   = (float*)(ws + o_Z);
    float* Htb  = (float*)(ws + o_Ht);
    float* gi   = (float*)(ws + o_xw2); // reuse after GCN GEMMs consumed
    float* hseq = (float*)(ws + o_Z);   // reuse after head computed
    float* hhd  = (float*)(ws + o_hh);
    float* ghd  = (float*)(ws + o_gh);
    float* deg  = (float*)(ws + o_deg);
    int*   srcA = (int*)(ws + o_src);
    int*   tgtA = (int*)(ws + o_tgt);
    float* nrmA = (float*)(ws + o_nrm);
    u16* Pg   = (u16*)(ws + o_Pg);
    u16* Plz  = (u16*)(ws + o_Plz);
    u16* Plh  = (u16*)(ws + o_Plh);
    u16* Pih0 = (u16*)(ws + o_Pih0);
    u16* Phh0 = (u16*)(ws + o_Phh0);
    u16* Pih1 = (u16*)(ws + o_Pih1);
    u16* Phh1 = (u16*)(ws + o_Phh1);

    // ---- GCN normalization ----
    k_init_deg<<<(NN + 255) / 256, 256, 0, stream>>>(deg);
    k_deg_accum<<<(E + 255) / 256, 256, 0, stream>>>(ei, ew, deg, E);
    k_make_norm<<<(nEdge + 255) / 256, 256, 0, stream>>>(ei, ew, deg, srcA, tgtA, nrmA, E);

    // ---- pack bf16 weight fragments ----
    auto packN = [](int K, int Nc) { return ((K >> 5) * (Nc >> 4) * 512 + 255) / 256; };
    k_pack<<<packN(128, 256), 256, 0, stream>>>(Wz,   Pg,   128, 256, 256, 1, 0);
    k_pack<<<packN(128, 256), 256, 0, stream>>>(Wh,   Pg,   128, 256, 256, 1, 16);
    k_pack<<<packN(256, 256), 256, 0, stream>>>(lzW,  Plz,  256, 256, 256, 1, 0);
    k_pack<<<packN(256, 256), 256, 0, stream>>>(lhW,  Plh,  256, 256, 256, 1, 0);
    k_pack<<<packN(128, 768), 256, 0, stream>>>(Wih0, Pih0, 128, 768, 1, 128, 0); // Wih0^T
    k_pack<<<packN(256, 768), 256, 0, stream>>>(Whh0, Phh0, 256, 768, 1, 256, 0); // Whh0^T
    k_pack<<<packN(256, 768), 256, 0, stream>>>(Wih1, Pih1, 256, 768, 1, 256, 0); // Wih1^T
    k_pack<<<packN(256, 768), 256, 0, stream>>>(Whh1, Phh1, 256, 768, 1, 256, 0); // Whh1^T

    // ---- GCN branch: xw2 = x @ [Wz|Wh] ----  (A contiguous: rs=0, rm=0)
    k_gemm<<<dim3(M / 64, 512 / 64), 256, 0, stream>>>(x, 0, 0, 128, 0, Pg, nullptr,
                                                       xw2, 128, 7, 512, 0);
    // agg = bias + scatter(norm * gather(xw2))
    k_init_agg<<<(unsigned)((M * 512 + 255) / 256), 256, 0, stream>>>(bz, bh, agg, M * 512);
    k_scatter<<<(unsigned)(((long)nEdge * BB * 128 + 255) / 256), 256, 0, stream>>>(
        xw2, srcA, tgtA, nrmA, agg, nEdge);
    // Z = sigmoid(agg[:, :256] @ lzW_top + lzb);  Ht = tanh(agg[:, 256:] @ lhW_top + lhb)
    k_gemm<<<dim3(M / 64, 256 / 64), 256, 0, stream>>>(agg, 0, 0, 512, 0, Plz, lzb,
                                                       Zb, 256, 8, 256, 1);
    k_gemm<<<dim3(M / 64, 256 / 64), 256, 0, stream>>>(agg + 256, 0, 0, 512, 0, Plh, lhb,
                                                       Htb, 256, 8, 256, 2);
    k_hn_head<<<(unsigned)(M / 8), 256, 0, stream>>>(Zb, Htb, l1W, l1b, hhd);

    // ---- GRU branch ----
    const size_t SMEM = 64 * 264 * 2 + 64 * 768 * 4; // 230400 B dynamic LDS
    // gi0 = xs @ Wih0^T + bih0  (xs row r = n*B + b: off = (r>>6)*CIN + (r&63)*N*CIN)
    k_gemm<<<dim3(M / 64, 768 / 64), 256, 0, stream>>>(x, 6, 63, CIN, (long)NN * CIN,
                                                       Pih0, bih0, gi, 128, 7, 768, 0);
    k_scan<<<1, 1024, SMEM, stream>>>(gi, Phh0, bhh0, hseq);
    // gi1 = h1 @ Wih1^T + bih1  (contiguous, stride 256)
    k_gemm<<<dim3(M / 64, 768 / 64), 256, 0, stream>>>(hseq, 0, 0, 256, 0,
                                                       Pih1, bih1, gi, 256, 8, 768, 0);
    k_scan<<<1, 1024, SMEM, stream>>>(gi, Phh1, bhh1, hseq);
    k_gru_head<<<(unsigned)(M / 8), 256, 0, stream>>>(hseq, l2W, l2b, ghd);

    // ---- final ----
    k_final<<<(unsigned)((M * OUTD + 255) / 256), 256, 0, stream>>>(
        ghd, hhd, l3W, l3b, (float*)d_out, M * OUTD);

    (void)n_in; (void)out_size; (void)ws_size; (void)in_sizes;
}
